// SplitBruteForceAntisymmetryWithDecay_3470333575740
// MI455X (gfx1250) — matrix operand order repaired and checked by reading, not compiled, not run on hardware
//
#include <hip/hip_runtime.h>
#include <math.h>

#define NELEC 10
#define NSPIN 2
#define NPER 5
#define DIMS 3
#define NION 2
#define DP 32
#define NDENSE 64
#define NPERM 120
#define NPERM_PAD 128
#define AST 40    // row stride (halfs) for K<=32 f16 tiles: 32 + 8 pad
#define H0S 72    // h0 row stride (halfs): 64 + 8 pad

// workspace layout (halfs): preconverted weight tiles, exact LDS image
#define WS_W0T_PER_SPIN (320 * AST)            // 12800 halfs
#define WS_W1T_PER_SPIN (NDENSE * NDENSE)      // 4096 halfs
#define WS_W1T_BASE     (NSPIN * WS_W0T_PER_SPIN)

typedef _Float16 v16h __attribute__((ext_vector_type(16)));
typedef float    v8f  __attribute__((ext_vector_type(8)));
typedef unsigned int u32x4 __attribute__((ext_vector_type(4)));
typedef int          i32x4 __attribute__((ext_vector_type(4)));
typedef int          i32x8 __attribute__((ext_vector_type(8)));

// branch-free tanh on the HW transcendental pipe: no exec-mask games
__device__ __forceinline__ float fast_tanh(float x) {
  float e = __builtin_amdgcn_exp2f(x * 2.885390081777927f); // exp(2x)
  return 1.0f - 2.0f * __builtin_amdgcn_rcpf(1.0f + e);
}

// TDM: contiguous global -> LDS copy of `ndwords` dwords (1-D tensor D#).
// Issue from ONE wave; tracked by TENSORcnt.
__device__ __forceinline__ void tdm_copy_to_lds(unsigned lds_off,
                                                const void* gptr,
                                                unsigned ndwords) {
  unsigned long long ga = (unsigned long long)gptr;
  u32x4 g0;
  g0[0] = 1u;                                          // count=1, user mode
  g0[1] = lds_off;                                     // lds_addr [63:32]
  g0[2] = (unsigned)(ga & 0xFFFFFFFFu);                // global_addr lo
  g0[3] = (unsigned)((ga >> 32) & 0x01FFFFFFu)         // global_addr hi
        | (2u << 30);                                  // type=2 ("image")
  i32x8 g1;
  g1[0] = 0x20000;                                     // data_size=4B code 2
  g1[1] = (int)((ndwords & 0xFFFFu) << 16);            // tensor_dim0 lo16
  g1[2] = (int)((ndwords >> 16) | (1u << 16));         // tensor_dim0 hi | tensor_dim1=1
  g1[3] = (int)((ndwords & 0xFFFFu) << 16);            // tile_dim0
  g1[4] = 1;                                           // tile_dim1=1
  g1[5] = (int)ndwords;                                // tensor_dim0_stride lo
  g1[6] = 0;
  g1[7] = 0;
  i32x4 z4 = {0, 0, 0, 0};
#if __clang_major__ >= 23
  i32x8 z8 = {0, 0, 0, 0, 0, 0, 0, 0};
  __builtin_amdgcn_tensor_load_to_lds(g0, g1, z4, z4, z8, 0);
#else
  __builtin_amdgcn_tensor_load_to_lds(g0, g1, z4, z4, 0);
#endif
}

struct SMem {
  union {
    _Float16 w0t[320 * AST];             // W0 re-indexed+transposed: [i*64+n][k]
    struct {
      _Float16 h0[NPERM_PAD * H0S];      // tanh activations (WMMA A for GEMM3)
      _Float16 w1t[NDENSE * NDENSE];     // W1 transposed [n][k]
    } g3;
  } u;
  _Float16 streamh[NSPIN][16 * AST];     // padded f16 backflow stream per spin
  _Float16 P[NPERM_PAD * AST];           // 0/1 permutation matrix [p][j*5+i]
  _Float16 Gt[NDENSE * AST];             // G transposed: [n][pair]
  float rn[NELEC * NION];
  float signs[NPERM_PAD];
  unsigned char perm[NPERM_PAD][8];
  float b0sh[NDENSE], b1sh[NDENSE], w2sh[NDENSE];
  float red[8];
  float ps[NSPIN];
};

// ---- prep: convert W0/W1 to f16 tile images in workspace (once/launch) ----
__global__ __launch_bounds__(256)
void prep_weights(const float* __restrict__ w0, const float* __restrict__ w1,
                  _Float16* __restrict__ ws) {
  int tid = blockIdx.x * 256 + threadIdx.x;
  const int total0 = NSPIN * NPER * DP * NDENSE;       // 20480
  if (tid < total0) {
    int s = tid / (NPER * DP * NDENSE);
    int idx = tid % (NPER * DP * NDENSE);
    int row = idx >> 6, n = idx & 63;                  // row = i*32+k
    int i = row >> 5, k = row & 31;
    ws[s * WS_W0T_PER_SPIN + (i * NDENSE + n) * AST + k] = (_Float16)w0[tid];
  } else if (tid < total0 + NSPIN * NDENSE * NDENSE) { // 8192 more
    int t = tid - total0;
    int s = t / (NDENSE * NDENSE);
    int idx = t % (NDENSE * NDENSE);
    int k = idx >> 6, n = idx & 63;
    ws[WS_W1T_BASE + s * WS_W1T_PER_SPIN + n * NDENSE + k] =
        (_Float16)w1[s * NDENSE * NDENSE + k * NDENSE + n];
  }
}

__global__ __launch_bounds__(256)
void wavefn_kernel(const float* __restrict__ elec_pos,
                   const float* __restrict__ ion_pos,
                   const float* __restrict__ bf_w,
                   const float* __restrict__ bf_b,
                   const float* __restrict__ b0,
                   const float* __restrict__ b1,
                   const float* __restrict__ w2,
                   const float* __restrict__ jk,
                   const _Float16* __restrict__ ws,
                   float* __restrict__ out)
{
  __shared__ SMem sm;
  const int tid  = threadIdx.x;
  const int b    = blockIdx.x;
  const int lane = tid & 31;
  const int wave = tid >> 5;
  const int nloc = lane & 15;   // tile column (N) / tile row (A-M) within wave
  const int hsel = lane >> 4;   // lane-half selector
  const float* ep = elec_pos + (size_t)b * NELEC * DIMS;

  const unsigned off_w0t = (unsigned)(unsigned long long)(const void*)&sm.u.w0t[0];
  const unsigned off_w1t = (unsigned)(unsigned long long)(const void*)&sm.u.g3.w1t[0];

  // ================= phase 0: r_norm + perms/signs =================
  if (tid < NELEC * NION) {
    int e = tid >> 1, i = tid & 1;
    float s = 0.f;
#pragma unroll
    for (int d = 0; d < DIMS; ++d) {
      float df = ep[e * DIMS + d] - ion_pos[i * DIMS + d];
      s += df * df;
    }
    sm.rn[tid] = sqrtf(s);
  }
  if (tid < NPERM_PAD) {
    if (tid < NPERM) {
      int rem = tid, inv = 0;
      const int fact[4] = {24, 6, 2, 1};
      bool used[5] = {false, false, false, false, false};
      for (int pos = 0; pos < NPER; ++pos) {
        int d = (pos < 4) ? (rem / fact[pos]) : 0;
        if (pos < 4) rem %= fact[pos];
        inv += d;
        int cnt = 0;
        for (int v = 0; v < NPER; ++v) {
          if (!used[v]) {
            if (cnt == d) { sm.perm[tid][pos] = (unsigned char)v; used[v] = true; break; }
            ++cnt;
          }
        }
      }
      sm.signs[tid] = (inv & 1) ? -1.f : 1.f;
    } else {
      sm.signs[tid] = 0.f;
      for (int pos = 0; pos < NPER; ++pos) sm.perm[tid][pos] = 0;
    }
  }
  __syncthreads();

  // ===== phase 1: padded f16 stream, P matrix, Gt K-pad zero =====
  for (int idx = tid; idx < NSPIN * 16 * AST; idx += 256) {
    int s = idx / (16 * AST);
    int r = (idx / AST) & 15;
    int c = idx % AST;
    float v = 0.f;
    if (r < NPER && c < DP) {
      int e = s * NPER + r;
      float feats[8];
#pragma unroll
      for (int i = 0; i < NION; ++i)
#pragma unroll
        for (int d = 0; d < DIMS; ++d)
          feats[i * DIMS + d] = ep[e * DIMS + d] - ion_pos[i * DIMS + d];
      feats[6] = sm.rn[e * 2 + 0];
      feats[7] = sm.rn[e * 2 + 1];
      float acc = bf_b[c];
#pragma unroll
      for (int f = 0; f < 8; ++f) acc += feats[f] * bf_w[f * DP + c];
      v = fast_tanh(acc);
    }
    sm.streamh[s][r * AST + c] = (_Float16)v;
  }
  for (int idx = tid; idx < NPERM_PAD * 32; idx += 256) {
    int p = idx >> 5, k = idx & 31;
    _Float16 v = (_Float16)0.f;
    if (p < NPERM && k < NPER * NPER) {
      int j = k / NPER, i = k % NPER;
      v = (sm.perm[p][i] == j) ? (_Float16)1.f : (_Float16)0.f;
    }
    sm.P[p * AST + k] = v;
  }
  for (int idx = tid; idx < NDENSE * 7; idx += 256) {
    int n = idx / 7, k = 25 + idx % 7;
    sm.Gt[n * AST + k] = (_Float16)0.f;
  }
  __syncthreads();

  // preload P A-fragment for this wave (shared across both spins)
  v16h pa;
  {
    const _Float16* prow = &sm.P[(wave * 16 + nloc) * AST];
    const int kb = hsel * 8;
#pragma unroll
    for (int e = 0; e < 8; ++e) {
      pa[e]     = prow[kb + e];
      pa[8 + e] = prow[16 + kb + e];
    }
  }

  for (int s = 0; s < NSPIN; ++s) {
    // ---- phase A: TDM-copy preconverted W0 tile image into LDS ----
    if (wave == 0)
      tdm_copy_to_lds(off_w0t, ws + (size_t)s * WS_W0T_PER_SPIN,
                      WS_W0T_PER_SPIN / 2);            // dwords
    if (tid < NDENSE) {
      sm.b0sh[tid] = b0[s * NDENSE + tid];
      sm.b1sh[tid] = b1[s * NDENSE + tid];
      sm.w2sh[tid] = w2[s * NDENSE + tid];
    }
    if (wave == 0) __builtin_amdgcn_s_wait_tensorcnt(0);
    __syncthreads();

    // ---- phase B: WMMA #1  G[j][i*64+n] = stream @ W0r ----
    {
      const _Float16* srow = &sm.streamh[s][nloc * AST];
      const int kb = hsel * 8;
      v16h sa;
#pragma unroll
      for (int e = 0; e < 8; ++e) {
        sa[e]     = srow[kb + e];
        sa[8 + e] = srow[16 + kb + e];
      }
      for (int nt = wave; nt < 20; nt += 8) {          // 320 cols / 16
        const _Float16* brow = &sm.u.w0t[(nt * 16 + nloc) * AST];
        const int k0 = hsel * 8;
        v16h bf;
#pragma unroll
        for (int e = 0; e < 16; ++e) bf[e] = brow[k0 + e + (e >= 8 ? 8 : 0)];
        v8f acc = {};
        acc = __builtin_amdgcn_wmma_f32_16x16x32_f16(false, sa, false, bf,
                                                     (short)0, acc, false, false);
        if (hsel == 0) {                               // rows j = v, j<5 real
          int Ng = nt * 16 + nloc;
          int i = Ng >> 6, n = Ng & 63;
#pragma unroll
          for (int v = 0; v < 5; ++v)
            sm.Gt[n * AST + (v * NPER + i)] = (_Float16)acc[v];
        }
      }
    }
    __syncthreads();

    // ---- phase C: issue W1^T TDM (overlaps WMMA #2), h0 = tanh(P@G+b0) ----
    if (wave == 0)
      tdm_copy_to_lds(off_w1t,
                      ws + WS_W1T_BASE + (size_t)s * WS_W1T_PER_SPIN,
                      WS_W1T_PER_SPIN / 2);            // dwords
#pragma unroll
    for (int nt = 0; nt < 4; ++nt) {
      const _Float16* brow = &sm.Gt[(nt * 16 + nloc) * AST];
      const int k0 = hsel * 8;
      v16h bf;
#pragma unroll
      for (int e = 0; e < 16; ++e) bf[e] = brow[k0 + e + (e >= 8 ? 8 : 0)];
      v8f acc = {};
      acc = __builtin_amdgcn_wmma_f32_16x16x32_f16(false, pa, false, bf,
                                                   (short)0, acc, false, false);
#pragma unroll
      for (int v = 0; v < 8; ++v) {
        int m = wave * 16 + hsel * 8 + v;
        int n = nt * 16 + nloc;
        float h = fast_tanh(acc[v] + sm.b0sh[n]);
        sm.u.g3.h0[m * H0S + n] = (m < NPERM) ? (_Float16)h : (_Float16)0.f;
      }
    }
    if (wave == 0) __builtin_amdgcn_s_wait_tensorcnt(0);
    __syncthreads();

    // ---- phase D: WMMA #3  h1 = tanh(h0@W1+b1)+h0, fused sign*W2 dot ----
    float local = 0.f;
    {
      const _Float16* arow = &sm.u.g3.h0[(wave * 16 + nloc) * H0S];
      const int kb = hsel * 8;
      v16h a0, a1;
#pragma unroll
      for (int e = 0; e < 8; ++e) {
        a0[e]     = arow[kb + e];
        a0[8 + e] = arow[kb + 16 + e];
        a1[e]     = arow[32 + kb + e];
        a1[8 + e] = arow[32 + kb + 16 + e];
      }
#pragma unroll
      for (int nt = 0; nt < 4; ++nt) {
        const _Float16* brow = &sm.u.g3.w1t[(nt * 16 + nloc) * NDENSE];
        const int k0 = hsel * 8;
        v16h bf0, bf1;
#pragma unroll
        for (int e = 0; e < 16; ++e) {
          bf0[e] = brow[k0 + e + (e >= 8 ? 8 : 0)];
          bf1[e] = brow[32 + k0 + e + (e >= 8 ? 8 : 0)];
        }
        v8f acc = {};
        acc = __builtin_amdgcn_wmma_f32_16x16x32_f16(false, a0, false, bf0,
                                                     (short)0, acc, false, false);
        acc = __builtin_amdgcn_wmma_f32_16x16x32_f16(false, a1, false, bf1,
                                                     (short)0, acc, false, false);
#pragma unroll
        for (int v = 0; v < 8; ++v) {
          int m = wave * 16 + hsel * 8 + v;
          int n = nt * 16 + nloc;
          float h1 = fast_tanh(acc[v] + sm.b1sh[n]) +
                     (float)sm.u.g3.h0[m * H0S + n];
          local += h1 * sm.signs[m] * sm.w2sh[n];
        }
      }
    }

    // ---- reduction: wave32 shuffle + tiny LDS combine ----
#pragma unroll
    for (int off = 16; off > 0; off >>= 1)
      local += __shfl_down(local, off, 32);
    if (lane == 0) sm.red[wave] = local;
    __syncthreads();
    if (tid == 0) {
      float t = 0.f;
#pragma unroll
      for (int w = 0; w < 8; ++w) t += sm.red[w];
      sm.ps[s] = t;
    }
    __syncthreads();
  }

  // ---- finalize: log|ps0*ps1| + jastrow ----
  if (tid == 0) {
    float jast = 0.f;
#pragma unroll
    for (int e = 0; e < NELEC; ++e)
#pragma unroll
      for (int i = 0; i < NION; ++i)
        jast -= jk[i] * sm.rn[e * NION + i];
    out[b] = logf(fabsf(sm.ps[0] * sm.ps[1])) + jast;
  }
}

extern "C" void kernel_launch(void* const* d_in, const int* in_sizes, int n_in,
                              void* d_out, int out_size, void* d_ws, size_t ws_size,
                              hipStream_t stream) {
  const int B = in_sizes[0] / (NELEC * DIMS);  // elec_pos is [B,10,3]
  _Float16* ws = (_Float16*)d_ws;              // needs ~68KB of scratch

  const int prep_elems = NSPIN * NPER * DP * NDENSE + NSPIN * NDENSE * NDENSE;
  prep_weights<<<(prep_elems + 255) / 256, 256, 0, stream>>>(
      (const float*)d_in[4],   // w0
      (const float*)d_in[6],   // w1
      ws);

  wavefn_kernel<<<B, 256, 0, stream>>>(
      (const float*)d_in[0],   // elec_pos
      (const float*)d_in[1],   // ion_pos
      (const float*)d_in[2],   // bf_w
      (const float*)d_in[3],   // bf_b
      (const float*)d_in[5],   // b0
      (const float*)d_in[7],   // b1
      (const float*)d_in[8],   // w2
      (const float*)d_in[10],  // jastrow_kernel
      ws,
      (float*)d_out);
}
